// HSSPPI_80882824118934
// MI455X (gfx1250) — matrix operand order repaired and checked by reading, not compiled, not run on hardware
//
#include <hip/hip_runtime.h>

// ---------------------------------------------------------------------------
// HSSPPI GNN for MI455X (gfx1250, wave32, WMMA).
//  - All GEMMs run through one WMMA kernel: A is f32 row-major, converted to
//    bf16 fragments in-register; B is pre-transposed bf16 [N,K]; f32 accum via
//    v_wmma_f32_16x16x32_bf16. Each wave owns a 16x64 output strip.
//  - "wide" mapping (N==256): one block covers 32 rows x 256 cols so the big
//    f32 adjacency (256MB, > L2) streams from HBM exactly once.
//  - GRU scans keep Wh (128x384 f32 = 192KB) resident in the 320KB WGP LDS.
//  - Segment-sum is deterministic (binary search on sorted map, no atomics).
// ---------------------------------------------------------------------------

typedef __bf16 bf16_t;
typedef __attribute__((ext_vector_type(16))) __bf16 v16bf;
typedef __attribute__((ext_vector_type(8)))  float  v8f;

enum { NA = 8192, NR = 1024, AIN = 7, RIN = 62, H1 = 256, H2 = 128, G3 = 384, MLP = 256 };

// ----------------------------- small utility kernels -----------------------

// Pad [M,K] f32 -> [M,Kp] f32 with zeros (Kp multiple of 32).
__global__ void k_pad(const float* __restrict__ X, float* __restrict__ Xp,
                      int M, int K, int Kp) {
  long i = (long)blockIdx.x * blockDim.x + threadIdx.x;
  if (i >= (long)M * Kp) return;
  int m = (int)(i / Kp), k = (int)(i % Kp);
  Xp[i] = (k < K) ? X[(long)m * K + k] : 0.0f;
}

// Transpose-and-convert weight W[K,N] f32 -> WT[N,Kp] bf16 (zero pad K..Kp).
__global__ void k_wtransT(const float* __restrict__ W, bf16_t* __restrict__ WT,
                          int K, int N, int Kp) {
  long i = (long)blockIdx.x * blockDim.x + threadIdx.x;
  if (i >= (long)N * Kp) return;
  int n = (int)(i / Kp), k = (int)(i % Kp);
  WT[i] = (k < K) ? (bf16_t)W[(long)k * N + n] : (bf16_t)0.0f;
}

__global__ void k_add(const float* __restrict__ a, const float* __restrict__ b,
                      float* __restrict__ c, int n) {
  int i = blockIdx.x * blockDim.x + threadIdx.x;
  if (i < n) c[i] = a[i] + b[i];
}

// dst[r,f] = rin[r,f] + sum_{a: map[a]==r} ain[a,f]   (map sorted ascending)
__global__ void k_seg_combine(const float* __restrict__ rin,
                              const float* __restrict__ ain,
                              const int* __restrict__ map,
                              float* __restrict__ dst,
                              int nAtoms, int nRes, int F) {
  int i = blockIdx.x * blockDim.x + threadIdx.x;
  if (i >= nRes * F) return;
  int r = i / F, f = i % F;
  int lo = 0, hi = nAtoms;                       // lower_bound(r)
  while (lo < hi) { int mid = (lo + hi) >> 1; if (map[mid] < r) lo = mid + 1; else hi = mid; }
  int s = lo;
  hi = nAtoms;                                   // lower_bound(r+1)
  while (lo < hi) { int mid = (lo + hi) >> 1; if (map[mid] < r + 1) lo = mid + 1; else hi = mid; }
  int e = lo;
  float acc = rin[i];
  for (int a = s; a < e; ++a) acc += ain[(long)a * F + f];
  dst[i] = acc;
}

// out[i] = sigmoid(dot(H[i,:], W2) + b2)
__global__ void k_head(const float* __restrict__ H, const float* __restrict__ W2,
                       const float* __restrict__ b2, float* __restrict__ out,
                       int M, int K) {
  int i = blockIdx.x * blockDim.x + threadIdx.x;
  if (i >= M) return;
  float acc = b2[0];
  for (int k = 0; k < K; ++k) acc += H[(long)i * K + k] * W2[k];
  out[i] = 1.0f / (1.0f + __expf(-acc));
}

// ----------------------------- WMMA GEMM ------------------------------------
// C[M,N] = act( A[M,K](f32) * B + bias ), B supplied transposed as BT[N,K] bf16.
// Block = 256 threads (8 waves), each wave owns a 16x64 output strip.
//   wide==0 (any N%64==0): block = 128 rows x 64 cols, grid=(N/64, M/128)
//   wide==1 (N==256 only): block =  32 rows x 256 cols, grid=(1, M/32)
//     -> grid.x==1 so a streamed A (the 256MB adjacency, bigger than L2) is
//        read from HBM exactly once; waves sharing an A tile hit the WGP$.
// A fragment (16-bit A 16x32): lane<16 -> K[0..16) of row lane%16;
// lane>=16 -> K[16..32). B fragment: lane reads 16 contiguous K of
// column n (low/high K-half by lane group) -> one 32-byte load from BT.
__global__ void k_gemm(const float* __restrict__ A,
                       const bf16_t* __restrict__ BT,
                       const float* __restrict__ bias,
                       float* __restrict__ outF,      // [M,N] f32 or null
                       bf16_t* __restrict__ outBfT,   // [N,M] bf16 or null
                       int M, int N, int K, int relu, int wide) {
  const int lane = threadIdx.x & 31;
  const int wave = threadIdx.x >> 5;
  const int half = lane >> 4;     // 0: low K half, 1: high K half
  const int lr   = lane & 15;
  int m0, n0;
  if (wide) {               // 8 waves = 2 M-tiles x 4 N-tiles, N must be 256
    m0 = blockIdx.y * 32 + (wave >> 2) * 16;
    n0 = (wave & 3) * 64;
  } else {                  // 8 waves stacked in M
    m0 = blockIdx.y * 128 + wave * 16;
    n0 = blockIdx.x * 64;
  }

  v8f acc[4] = {};
  const float* arow = A + (size_t)(m0 + lr) * K;

  for (int k = 0; k < K; k += 32) {
    const float* p0 = arow + k + half * 16;
    const float* p1 = arow + k + half * 16 + 8;
    float4 f0 = *(const float4*)(p0);
    float4 f1 = *(const float4*)(p0 + 4);
    float4 f2 = *(const float4*)(p1);
    float4 f3 = *(const float4*)(p1 + 4);
    // Speculative prefetch of the next A chunk (dropped silently past the end).
    __builtin_prefetch(arow + k + 64, 0, 3);

    v16bf av;
    av[0]  = (bf16_t)f0.x; av[1]  = (bf16_t)f0.y; av[2]  = (bf16_t)f0.z; av[3]  = (bf16_t)f0.w;
    av[4]  = (bf16_t)f1.x; av[5]  = (bf16_t)f1.y; av[6]  = (bf16_t)f1.z; av[7]  = (bf16_t)f1.w;
    av[8]  = (bf16_t)f2.x; av[9]  = (bf16_t)f2.y; av[10] = (bf16_t)f2.z; av[11] = (bf16_t)f2.w;
    av[12] = (bf16_t)f3.x; av[13] = (bf16_t)f3.y; av[14] = (bf16_t)f3.z; av[15] = (bf16_t)f3.w;

#pragma unroll
    for (int j = 0; j < 4; ++j) {
      v16bf bv = *(const v16bf*)(BT + (size_t)(n0 + j * 16 + lr) * K + k + half * 16);
      acc[j] = __builtin_amdgcn_wmma_f32_16x16x32_bf16(
          /*neg_a=*/false, av, /*neg_b=*/false, bv,
          /*c_mod=*/(short)0, acc[j], /*reuse_a=*/false, /*reuse_b=*/false);
    }
  }

#pragma unroll
  for (int j = 0; j < 4; ++j) {
    int n = n0 + j * 16 + lr;
    float bv = bias ? bias[n] : 0.0f;
#pragma unroll
    for (int v = 0; v < 8; ++v) {
      int m = m0 + half * 8 + v;
      float d = acc[j][v] + bv;
      if (relu) d = fmaxf(d, 0.0f);
      if (outF)   outF[(size_t)m * N + n] = d;
      if (outBfT) outBfT[(size_t)n * M + m] = (bf16_t)d;
    }
  }
}

// ----------------------------- GRU scan -------------------------------------
// One block per direction (gridDim.x == 2), 384 threads. Wh (128x384 f32,
// 192KB) + h(128) + hp(384) live in the 320KB WGP LDS. Two barriers per step.
__global__ void k_gru_scan(const float* __restrict__ xp0, const float* __restrict__ xp1,
                           const float* __restrict__ Wh0, const float* __restrict__ Wh1,
                           const float* __restrict__ bh0, const float* __restrict__ bh1,
                           float* __restrict__ out, int N) {
  extern __shared__ float smem[];
  float* sWh = smem;                 // [128*384]
  float* sH  = smem + 128 * 384;     // [128]
  float* sHp = sH + 128;             // [384]

  const int dir = blockIdx.x;
  const float* xp = dir ? xp1 : xp0;
  const float* Wh = dir ? Wh1 : Wh0;
  const float* bh = dir ? bh1 : bh0;
  const int tid = threadIdx.x;       // 0..383

  for (int i = tid; i < 128 * 384; i += 384) sWh[i] = Wh[i];
  if (tid < 128) sH[tid] = 0.0f;
  __syncthreads();

  const float bhv = bh[tid];
  for (int t = 0; t < N; ++t) {
    const int row = dir ? (N - 1 - t) : t;
    const float* xr = xp + (size_t)row * 384;

    float acc = bhv;                 // hp = h @ Wh + bh  (column tid)
#pragma unroll 8
    for (int i = 0; i < 128; ++i) acc += sH[i] * sWh[i * 384 + tid];
    sHp[tid] = acc;
    __syncthreads();

    if (tid < 128) {
      float r = 1.0f / (1.0f + __expf(-(xr[tid]       + sHp[tid])));
      float z = 1.0f / (1.0f + __expf(-(xr[128 + tid] + sHp[128 + tid])));
      float n = tanhf(xr[256 + tid] + r * sHp[256 + tid]);
      float hn = (1.0f - z) * n + z * sH[tid];
      sH[tid] = hn;
      out[(size_t)row * 256 + dir * 128 + tid] = hn;
    }
    __syncthreads();
  }
}

// ----------------------------- host-side orchestration ----------------------

static inline void* wsoff(void* ws, size_t& off, size_t bytes) {
  char* p = (char*)ws + off;
  off += (bytes + 255) & ~(size_t)255;
  return p;
}

struct Scratch {
  float* padA; bf16_t* BT; bf16_t* xwT; float* hbuf; float* xp0; float* xp1;
};

// One GCN + biGRU block: o[Nb,256] = block(x[Nb,Din], adj[Nb,Nb], weights[bidx])
static void run_block(hipStream_t s, int Nb, const float* x, int Din,
                      const float* adj, const float* Wg, const float* bgv,
                      const float* gWx, const float* gWh,
                      const float* gbx, const float* gbh, int bidx,
                      float* o, const Scratch& sc) {
  int Kp = (Din + 31) & ~31;
  const float* Ax = x;
  if (Kp != Din) {
    long tot = (long)Nb * Kp;
    k_pad<<<dim3((unsigned)((tot + 255) / 256)), 256, 0, s>>>(x, sc.padA, Nb, Din, Kp);
    Ax = sc.padA;
  }
  // xw = x @ Wg  -> stored transposed bf16 [256, Nb] (B operand of adj GEMM)
  {
    long tot = (long)H1 * Kp;
    k_wtransT<<<dim3((unsigned)((tot + 255) / 256)), 256, 0, s>>>(Wg, sc.BT, Din, H1, Kp);
  }
  k_gemm<<<dim3(1, Nb / 32), 256, 0, s>>>(Ax, sc.BT, nullptr,
                                          nullptr, sc.xwT, Nb, H1, Kp, 0, /*wide=*/1);
  // h = relu(adj @ xw + bg)   (adj f32 -> bf16 in-register; wide: HBM-once)
  k_gemm<<<dim3(1, Nb / 32), 256, 0, s>>>(adj, sc.xwT, bgv,
                                          sc.hbuf, nullptr, Nb, H1, Nb, 1, /*wide=*/1);
  const float* Wx0 = gWx + (size_t)(bidx * 2 + 0) * H1 * G3;
  const float* Wx1 = gWx + (size_t)(bidx * 2 + 1) * H1 * G3;
  const float* Wh0 = gWh + (size_t)(bidx * 2 + 0) * H2 * G3;
  const float* Wh1 = gWh + (size_t)(bidx * 2 + 1) * H2 * G3;
  const float* bx0 = gbx + (bidx * 2 + 0) * G3;
  const float* bx1 = gbx + (bidx * 2 + 1) * G3;
  const float* bh0 = gbh + (bidx * 2 + 0) * G3;
  const float* bh1 = gbh + (bidx * 2 + 1) * G3;
  // xp[d] = h @ Wx[d] + bx[d]   (N=384 -> narrow mapping)
  long wt = (long)G3 * H1;
  k_wtransT<<<dim3((unsigned)((wt + 255) / 256)), 256, 0, s>>>(Wx0, sc.BT, H1, G3, H1);
  k_gemm<<<dim3(G3 / 64, Nb / 128), 256, 0, s>>>(sc.hbuf, sc.BT, bx0,
                                                 sc.xp0, nullptr, Nb, G3, H1, 0, 0);
  k_wtransT<<<dim3((unsigned)((wt + 255) / 256)), 256, 0, s>>>(Wx1, sc.BT, H1, G3, H1);
  k_gemm<<<dim3(G3 / 64, Nb / 128), 256, 0, s>>>(sc.hbuf, sc.BT, bx1,
                                                 sc.xp1, nullptr, Nb, G3, H1, 0, 0);
  // bidirectional GRU scan (fwd = block 0, bwd = block 1)
  size_t smem = (size_t)(128 * 384 + 128 + 384) * sizeof(float);
  k_gru_scan<<<2, 384, smem, s>>>(sc.xp0, sc.xp1, Wh0, Wh1, bh0, bh1, o, Nb);
}

extern "C" void kernel_launch(void* const* d_in, const int* in_sizes, int n_in,
                              void* d_out, int out_size, void* d_ws, size_t ws_size,
                              hipStream_t stream) {
  (void)in_sizes; (void)n_in; (void)out_size; (void)ws_size;
  const float* an    = (const float*)d_in[0];   // [NA,7]
  const float* aadj  = (const float*)d_in[1];   // [NA,NA]
  const float* rn    = (const float*)d_in[2];   // [NR,62]
  const float* radj  = (const float*)d_in[3];   // [NR,NR]
  const int*   map   = (const int*)  d_in[4];   // [NA] sorted
  const float* Wg_a1 = (const float*)d_in[5];
  const float* Wg_r1 = (const float*)d_in[6];
  const float* Wg_a2 = (const float*)d_in[7];
  const float* Wg_r2 = (const float*)d_in[8];
  const float* bg    = (const float*)d_in[9];   // [4,256]
  const float* gWx   = (const float*)d_in[10];  // [4,2,256,384]
  const float* gWh   = (const float*)d_in[11];  // [4,2,128,384]
  const float* gbx   = (const float*)d_in[12];  // [4,2,384]
  const float* gbh   = (const float*)d_in[13];  // [4,2,384]
  const float* W1    = (const float*)d_in[14];  // [256,256]
  const float* b1    = (const float*)d_in[15];  // [256]
  const float* W2    = (const float*)d_in[16];  // [256,1]
  const float* b2    = (const float*)d_in[17];  // [1]
  float* out = (float*)d_out;                   // [NR]

  size_t off = 0;
  Scratch sc;
  sc.padA = (float*) wsoff(d_ws, off, (size_t)NA * 64 * sizeof(float));
  sc.BT   = (bf16_t*)wsoff(d_ws, off, (size_t)G3 * H1 * sizeof(bf16_t));
  sc.xwT  = (bf16_t*)wsoff(d_ws, off, (size_t)H1 * NA * sizeof(bf16_t));
  sc.hbuf = (float*) wsoff(d_ws, off, (size_t)NA * H1 * sizeof(float));
  sc.xp0  = (float*) wsoff(d_ws, off, (size_t)NA * G3 * sizeof(float));
  sc.xp1  = (float*) wsoff(d_ws, off, (size_t)NA * G3 * sizeof(float));
  float* oa   = (float*)wsoff(d_ws, off, (size_t)NA * 256 * sizeof(float)); // a1 then a2
  float* orr  = (float*)wsoff(d_ws, off, (size_t)NR * 256 * sizeof(float)); // r1 then r2
  float* ar1  = (float*)wsoff(d_ws, off, (size_t)NR * 256 * sizeof(float));
  float* ar2  = (float*)wsoff(d_ws, off, (size_t)NR * 256 * sizeof(float));
  float* skip = (float*)wsoff(d_ws, off, (size_t)NR * 256 * sizeof(float));
  float* hmlp = (float*)wsoff(d_ws, off, (size_t)NR * 256 * sizeof(float));

  // a1 = block0(a_input), r1 = block1(r_input)
  run_block(stream, NA, an, AIN, aadj, Wg_a1, bg + 0 * H1, gWx, gWh, gbx, gbh, 0, oa,  sc);
  run_block(stream, NR, rn, RIN, radj, Wg_r1, bg + 1 * H1, gWx, gWh, gbx, gbh, 1, orr, sc);

  // ar1 = r1 + segment_sum(a1)
  k_seg_combine<<<dim3(NR * 256 / 256), 256, 0, stream>>>(orr, oa, map, ar1, NA, NR, 256);

  // a2 = block2(a1)  (oa is read fully by the first GEMM before being
  //  overwritten by the scans -> safe in-place reuse on the ordered stream)
  run_block(stream, NA, oa, 256, aadj, Wg_a2, bg + 2 * H1, gWx, gWh, gbx, gbh, 2, oa,  sc);
  // r2 = block3(ar1)
  run_block(stream, NR, ar1, 256, radj, Wg_r2, bg + 3 * H1, gWx, gWh, gbx, gbh, 3, orr, sc);

  // ar2 = r2 + segment_sum(a2);  skip = ar1 + ar2
  k_seg_combine<<<dim3(NR * 256 / 256), 256, 0, stream>>>(orr, oa, map, ar2, NA, NR, 256);
  k_add<<<dim3(NR * 256 / 256), 256, 0, stream>>>(ar1, ar2, skip, NR * 256);

  // h = relu(skip @ W1 + b1)  (WMMA, wide), out = sigmoid(h @ W2 + b2)
  long wt = (long)MLP * 256;
  k_wtransT<<<dim3((unsigned)((wt + 255) / 256)), 256, 0, stream>>>(W1, sc.BT, 256, MLP, 256);
  k_gemm<<<dim3(1, NR / 32), 256, 0, stream>>>(skip, sc.BT, b1,
                                               hmlp, nullptr, NR, MLP, 256, 1, /*wide=*/1);
  k_head<<<dim3(NR / 256), 256, 0, stream>>>(hmlp, W2, b2, out, NR, MLP);
}
